// BiLSTM_CRF_24489903522600
// MI455X (gfx1250) — matrix operand order repaired and checked by reading, not compiled, not run on hardware
//
#include <hip/hip_runtime.h>
#include <hip/hip_bf16.h>
#include <stdint.h>

// ---------------------------------------------------------------------------
// Types for WMMA operands (CDNA5 wave32)
// ---------------------------------------------------------------------------
typedef __attribute__((ext_vector_type(8)))  float  v8f;
typedef __attribute__((ext_vector_type(8)))  __bf16 v8bf;
typedef __attribute__((ext_vector_type(16))) __bf16 v16bf;

#define S_LEN 512
#define BATCH 64
#define EDIM  512
#define HDIM  512
#define GDIM  2048   // 4 * HDIM
#define NTAG  9

// Packed-weight geometry: K = EDIM + HDIM = 1024 -> 32 K-tiles of 32
//                         N = GDIM = 2048        -> 128 N-tiles of 16
#define KT_TOT 32
#define NT_TOT 128
#define PW_PER_DIR ((size_t)KT_TOT * NT_TOT * 512)   // bf16 elems = 2,097,152

union BFU { unsigned short u; __bf16 b; };

static __device__ __forceinline__ __bf16 f2bf(float f) {
  unsigned u = __float_as_uint(f);
  unsigned r = (u + 0x7fffu + ((u >> 16) & 1u)) >> 16;   // RNE
  BFU x; x.u = (unsigned short)r; return x.b;
}
static __device__ __forceinline__ float bf2f(__bf16 b) {
  BFU x; x.b = b; return __uint_as_float(((unsigned)x.u) << 16);
}

static __device__ __forceinline__ v8f wmma_bf16(v16bf a, v16bf b, v8f c) {
  return __builtin_amdgcn_wmma_f32_16x16x32_bf16(
      false, a, false, b, (short)0, c, false, false);
}

// A-operand (16x32 bf16): per lane two contiguous 8-element K chunks:
//   K = kb + hi*8 .. +7   and   K = kb + 16 + hi*8 .. +7
static __device__ __forceinline__ v16bf loadA(const __bf16* p0, const __bf16* p1) {
  v8bf a0 = *(const v8bf*)p0;
  v8bf a1 = *(const v8bf*)p1;
  v16bf a;
#pragma unroll
  for (int r = 0; r < 8; ++r) { a[r] = a0[r]; a[8 + r] = a1[r]; }
  return a;
}

static __device__ __forceinline__ v16bf loadB(const __bf16* pwd, int ktg, int nt, int lane) {
  return *(const v16bf*)(pwd + ((size_t)ktg * NT_TOT + nt) * 512 + (size_t)lane * 16);
}

static __device__ __forceinline__ float sigm(float x) {
  return 1.0f / (1.0f + __expf(-x));
}

// ---------------------------------------------------------------------------
// Kernel 1: pack combined [W_ih ; W_hh]^T (K=1024, N=2048) into WMMA B layout
// ---------------------------------------------------------------------------
__global__ void pack_weights_kernel(const float* __restrict__ wih_f,
                                    const float* __restrict__ whh_f,
                                    const float* __restrict__ wih_b,
                                    const float* __restrict__ whh_b,
                                    __bf16* __restrict__ pw) {
  size_t idx = (size_t)blockIdx.x * blockDim.x + threadIdx.x;
  if (idx >= 2 * PW_PER_DIR) return;
  int dir = (int)(idx / PW_PER_DIR);
  size_t r = idx % PW_PER_DIR;
  int ti   = (int)(r >> 9);
  int lj   = (int)(r & 511);
  int lane = lj >> 4;
  int j    = lj & 15;
  int kt   = ti >> 7;
  int nt   = ti & 127;
  int col   = lane & 15;
  int khalf = lane >> 4;
  int k = kt * 32 + khalf * 8 + (j & 7) + (j >> 3) * 16;
  int n = nt * 16 + col;
  const float* wih = dir ? wih_b : wih_f;
  const float* whh = dir ? whh_b : whh_f;
  float v = (k < EDIM) ? wih[(size_t)n * EDIM + k]
                       : whh[(size_t)n * HDIM + (k - EDIM)];
  pw[idx] = f2bf(v);
}

// ---------------------------------------------------------------------------
// Kernel 1b: pack w_lin (9 x 1024) into one 16-wide WMMA B column, zero-padded
// ---------------------------------------------------------------------------
__global__ void pack_wlin_kernel(const float* __restrict__ wlin,
                                 __bf16* __restrict__ plin) {
  int idx = blockIdx.x * blockDim.x + threadIdx.x;
  if (idx >= 32 * 512) return;
  int kt   = idx >> 9;
  int lj   = idx & 511;
  int lane = lj >> 4;
  int j    = lj & 15;
  int col   = lane & 15;
  int khalf = lane >> 4;
  int k = kt * 32 + khalf * 8 + (j & 7) + (j >> 3) * 16;
  float v = (col < NTAG) ? wlin[(size_t)col * (2 * HDIM) + k] : 0.0f;
  plin[idx] = f2bf(v);
}

// ---------------------------------------------------------------------------
// Kernel 2: embedding gather + cast -> x_bf16 (S, B, E) time-major
// ---------------------------------------------------------------------------
__global__ void embed_cast_kernel(const int* __restrict__ sentence,
                                  const float* __restrict__ emb,
                                  __bf16* __restrict__ x) {
  size_t idx = (size_t)blockIdx.x * blockDim.x + threadIdx.x;
  if (idx >= (size_t)S_LEN * BATCH * EDIM) return;
  int s   = (int)(idx >> 15);
  int rem = (int)(idx & 32767);
  int b   = rem >> 9;
  int e   = rem & 511;
  int tok = sentence[b * S_LEN + s];
  x[idx] = f2bf(emb[(size_t)tok * EDIM + e]);
}

// ---------------------------------------------------------------------------
// Kernel 3: pre-gates GEMM (whole-GPU parallel): pg[dir][t*64+b][n] =
//   bias[n] + sum_e x[t,b,e] * W_ih[n,e]     (M=32768, N=2048, K=512)
// 8 waves/block; each wave: 1 m-tile x 8 n-tiles, software-pipelined K loop.
// ---------------------------------------------------------------------------
__global__ __launch_bounds__(256)
void pregate_kernel(const __bf16* __restrict__ x, const __bf16* __restrict__ pw,
                    const float* __restrict__ bih_f, const float* __restrict__ bhh_f,
                    const float* __restrict__ bih_b, const float* __restrict__ bhh_b,
                    __bf16* __restrict__ pg) {
  const int lane = threadIdx.x & 31;
  const int laneRow = lane & 15, laneHi = lane >> 4;
  int w = blockIdx.x * 8 + (threadIdx.x >> 5);
  const int dir   = w >> 15;
  const int w2    = w & 32767;
  const int mtile = w2 >> 4;
  const int ngrp  = w2 & 15;        // 8 n-tiles each
  const __bf16* pwd = pw + (size_t)dir * PW_PER_DIR;
  const float* bih = dir ? bih_b : bih_f;
  const float* bhh = dir ? bhh_b : bhh_f;

  v8f acc[8];
#pragma unroll
  for (int q = 0; q < 8; ++q) {
    int n = (ngrp * 8 + q) * 16 + laneRow;
    float bv = bih[n] + bhh[n];
#pragma unroll
    for (int r = 0; r < 8; ++r) acc[q][r] = bv;
  }

  const __bf16* arow = x + (size_t)(mtile * 16 + laneRow) * EDIM;
  v16bf a = loadA(arow + laneHi * 8, arow + 16 + laneHi * 8);
  v16bf b[8];
#pragma unroll
  for (int q = 0; q < 8; ++q) b[q] = loadB(pwd, 0, ngrp * 8 + q, lane);

  for (int kt = 0; kt < 16; ++kt) {
    v16bf an = a, bn[8];
#pragma unroll
    for (int q = 0; q < 8; ++q) bn[q] = b[q];
    if (kt < 15) {
      int kb = (kt + 1) * 32;
      an = loadA(arow + kb + laneHi * 8, arow + kb + 16 + laneHi * 8);
#pragma unroll
      for (int q = 0; q < 8; ++q) bn[q] = loadB(pwd, kt + 1, ngrp * 8 + q, lane);
    }
#pragma unroll
    for (int q = 0; q < 8; ++q) acc[q] = wmma_bf16(a, b[q], acc[q]);
    a = an;
#pragma unroll
    for (int q = 0; q < 8; ++q) b[q] = bn[q];
  }

  __bf16* pgd = pg + (size_t)dir * S_LEN * BATCH * GDIM;
#pragma unroll
  for (int q = 0; q < 8; ++q) {
    int n = (ngrp * 8 + q) * 16 + laneRow;
#pragma unroll
    for (int r = 0; r < 8; ++r) {
      int row = mtile * 16 + r + laneHi * 8;
      pgd[(size_t)row * GDIM + n] = f2bf(acc[q][r]);
    }
  }
}

// ---------------------------------------------------------------------------
// Kernel 4a: recurrent LSTM with precomputed pre-gates.  grid = 2, block = 1024.
// Per step: acc init from pg, then only h @ W_hh^T (16 pipelined K-iters).
// ---------------------------------------------------------------------------
__global__ __launch_bounds__(1024)
void lstm_pg_kernel(const float* __restrict__ h0, const float* __restrict__ c0,
                    const __bf16* __restrict__ pw, const __bf16* __restrict__ pg,
                    __bf16* __restrict__ hs) {
  const int dir  = blockIdx.x;
  const int tid  = threadIdx.x;
  const int lane = tid & 31;
  const int wave = tid >> 5;
  const int laneRow = lane & 15, laneHi = lane >> 4;

  __shared__ __bf16 hb[BATCH * HDIM];   // 64 KB
  for (int i = tid; i < BATCH * HDIM; i += 1024)
    hb[i] = f2bf(h0[(size_t)dir * BATCH * HDIM + i]);

  const __bf16* pwd = pw + (size_t)dir * PW_PER_DIR;
  const __bf16* pgd = pg + (size_t)dir * S_LEN * BATCH * GDIM;

  float cst[4][8];
  int mArr[4], nhArr[4];
#pragma unroll
  for (int p4 = 0; p4 < 4; ++p4) {
    int pos = wave * 4 + p4;
    int m = pos >> 5, nh = pos & 31;
    mArr[p4] = m; nhArr[p4] = nh;
#pragma unroll
    for (int r = 0; r < 8; ++r) {
      int row  = m * 16 + r + laneHi * 8;
      int colh = nh * 16 + laneRow;
      cst[p4][r] = c0[(size_t)dir * BATCH * HDIM + (size_t)row * HDIM + colh];
    }
  }
  __syncthreads();

  for (int st = 0; st < S_LEN; ++st) {
    const int t = dir ? (S_LEN - 1 - st) : st;
    const int tn = dir ? (t - 1) : (t + 1);
    if (tn >= 0 && tn < S_LEN)   // prefetch next step's pre-gates (global_prefetch_b8)
      __builtin_prefetch(pgd + (size_t)tn * BATCH * GDIM + (size_t)tid * 128, 0, 1);

    float hstash[4][8];
    for (int p4 = 0; p4 < 4; ++p4) {
      const int m = mArr[p4], nh = nhArr[p4];
      v8f acc0, acc1, acc2, acc3;
      {
        const __bf16* pgbase = pgd + (size_t)(t * BATCH) * GDIM;
#pragma unroll
        for (int r = 0; r < 8; ++r) {
          int row = m * 16 + r + laneHi * 8;
          const __bf16* prow = pgbase + (size_t)row * GDIM + nh * 16 + laneRow;
          acc0[r] = bf2f(prow[0 * HDIM]);
          acc1[r] = bf2f(prow[1 * HDIM]);
          acc2[r] = bf2f(prow[2 * HDIM]);
          acc3[r] = bf2f(prow[3 * HDIM]);
        }
      }
      const __bf16* hrow = &hb[(size_t)(m * 16 + laneRow) * HDIM];
      v16bf a  = loadA(hrow + laneHi * 8, hrow + 16 + laneHi * 8);
      v16bf b0 = loadB(pwd, 16, 0 * 32 + nh, lane);
      v16bf b1 = loadB(pwd, 16, 1 * 32 + nh, lane);
      v16bf b2 = loadB(pwd, 16, 2 * 32 + nh, lane);
      v16bf b3 = loadB(pwd, 16, 3 * 32 + nh, lane);
      for (int kt = 0; kt < 16; ++kt) {
        v16bf an = a, c0v = b0, c1v = b1, c2v = b2, c3v = b3;
        if (kt < 15) {
          int kb = (kt + 1) * 32;
          an  = loadA(hrow + kb + laneHi * 8, hrow + kb + 16 + laneHi * 8);
          c0v = loadB(pwd, 17 + kt, 0 * 32 + nh, lane);
          c1v = loadB(pwd, 17 + kt, 1 * 32 + nh, lane);
          c2v = loadB(pwd, 17 + kt, 2 * 32 + nh, lane);
          c3v = loadB(pwd, 17 + kt, 3 * 32 + nh, lane);
        }
        acc0 = wmma_bf16(a, b0, acc0);
        acc1 = wmma_bf16(a, b1, acc1);
        acc2 = wmma_bf16(a, b2, acc2);
        acc3 = wmma_bf16(a, b3, acc3);
        a = an; b0 = c0v; b1 = c1v; b2 = c2v; b3 = c3v;
      }
#pragma unroll
      for (int r = 0; r < 8; ++r) {
        float iv = sigm(acc0[r]);
        float fv = sigm(acc1[r]);
        float gv = tanhf(acc2[r]);
        float ov = sigm(acc3[r]);
        float cn = fv * cst[p4][r] + iv * gv;
        cst[p4][r] = cn;
        float hv = ov * tanhf(cn);
        hstash[p4][r] = hv;
        int row  = m * 16 + r + laneHi * 8;
        int colh = nh * 16 + laneRow;
        hs[((size_t)t * BATCH + row) * (2 * HDIM) + (size_t)dir * HDIM + colh] = f2bf(hv);
      }
    }
    __syncthreads();
#pragma unroll
    for (int p4 = 0; p4 < 4; ++p4) {
      const int m = mArr[p4], nh = nhArr[p4];
#pragma unroll
      for (int r = 0; r < 8; ++r) {
        int row  = m * 16 + r + laneHi * 8;
        int colh = nh * 16 + laneRow;
        hb[(size_t)row * HDIM + colh] = f2bf(hstash[p4][r]);
      }
    }
    __syncthreads();
  }
}

// ---------------------------------------------------------------------------
// Kernel 4b: fused fallback (x-GEMM inside the step loop), software-pipelined.
// ---------------------------------------------------------------------------
__global__ __launch_bounds__(1024)
void lstm_fused_kernel(const float* __restrict__ h0, const float* __restrict__ c0,
                       const float* __restrict__ bih_f, const float* __restrict__ bhh_f,
                       const float* __restrict__ bih_b, const float* __restrict__ bhh_b,
                       const __bf16* __restrict__ pw, const __bf16* __restrict__ x,
                       __bf16* __restrict__ hs) {
  const int dir  = blockIdx.x;
  const int tid  = threadIdx.x;
  const int lane = tid & 31;
  const int wave = tid >> 5;
  const int laneRow = lane & 15, laneHi = lane >> 4;

  __shared__ __bf16 hb[BATCH * HDIM];
  for (int i = tid; i < BATCH * HDIM; i += 1024)
    hb[i] = f2bf(h0[(size_t)dir * BATCH * HDIM + i]);

  const __bf16* pwd = pw + (size_t)dir * PW_PER_DIR;
  const float* bih = dir ? bih_b : bih_f;
  const float* bhh = dir ? bhh_b : bhh_f;

  float cst[4][8];
  float bias[4][4];
  int mArr[4], nhArr[4];
#pragma unroll
  for (int p4 = 0; p4 < 4; ++p4) {
    int pos = wave * 4 + p4;
    int m = pos >> 5, nh = pos & 31;
    mArr[p4] = m; nhArr[p4] = nh;
#pragma unroll
    for (int g = 0; g < 4; ++g) {
      int n = g * HDIM + nh * 16 + laneRow;
      bias[p4][g] = bih[n] + bhh[n];
    }
#pragma unroll
    for (int r = 0; r < 8; ++r) {
      int row  = m * 16 + r + laneHi * 8;
      int colh = nh * 16 + laneRow;
      cst[p4][r] = c0[(size_t)dir * BATCH * HDIM + (size_t)row * HDIM + colh];
    }
  }
  __syncthreads();

  for (int st = 0; st < S_LEN; ++st) {
    const int t = dir ? (S_LEN - 1 - st) : st;
    const int tn = dir ? (t - 1) : (t + 1);
    const __bf16* xt = x + (size_t)t * BATCH * EDIM;
    if (tn >= 0 && tn < S_LEN)
      __builtin_prefetch(x + (size_t)tn * BATCH * EDIM + (size_t)tid * 32, 0, 1);

    float hstash[4][8];
    for (int p4 = 0; p4 < 4; ++p4) {
      const int m = mArr[p4], nh = nhArr[p4];
      v8f acc0, acc1, acc2, acc3;
#pragma unroll
      for (int r = 0; r < 8; ++r) {
        acc0[r] = bias[p4][0]; acc1[r] = bias[p4][1];
        acc2[r] = bias[p4][2]; acc3[r] = bias[p4][3];
      }
      const __bf16* arow = xt + (size_t)(m * 16 + laneRow) * EDIM;
      const __bf16* hrow = &hb[(size_t)(m * 16 + laneRow) * HDIM];
      v16bf a  = loadA(arow + laneHi * 8, arow + 16 + laneHi * 8);
      v16bf b0 = loadB(pwd, 0, 0 * 32 + nh, lane);
      v16bf b1 = loadB(pwd, 0, 1 * 32 + nh, lane);
      v16bf b2 = loadB(pwd, 0, 2 * 32 + nh, lane);
      v16bf b3 = loadB(pwd, 0, 3 * 32 + nh, lane);
      for (int kk = 0; kk < 32; ++kk) {
        v16bf an = a, c0v = b0, c1v = b1, c2v = b2, c3v = b3;
        if (kk < 31) {
          int nk = kk + 1;
          if (nk < 16) {
            int kb = nk * 32;
            an = loadA(arow + kb + laneHi * 8, arow + kb + 16 + laneHi * 8);
          } else {
            int kb = (nk - 16) * 32;
            an = loadA(hrow + kb + laneHi * 8, hrow + kb + 16 + laneHi * 8);
          }
          c0v = loadB(pwd, nk, 0 * 32 + nh, lane);
          c1v = loadB(pwd, nk, 1 * 32 + nh, lane);
          c2v = loadB(pwd, nk, 2 * 32 + nh, lane);
          c3v = loadB(pwd, nk, 3 * 32 + nh, lane);
        }
        acc0 = wmma_bf16(a, b0, acc0);
        acc1 = wmma_bf16(a, b1, acc1);
        acc2 = wmma_bf16(a, b2, acc2);
        acc3 = wmma_bf16(a, b3, acc3);
        a = an; b0 = c0v; b1 = c1v; b2 = c2v; b3 = c3v;
      }
#pragma unroll
      for (int r = 0; r < 8; ++r) {
        float iv = sigm(acc0[r]);
        float fv = sigm(acc1[r]);
        float gv = tanhf(acc2[r]);
        float ov = sigm(acc3[r]);
        float cn = fv * cst[p4][r] + iv * gv;
        cst[p4][r] = cn;
        float hv = ov * tanhf(cn);
        hstash[p4][r] = hv;
        int row  = m * 16 + r + laneHi * 8;
        int colh = nh * 16 + laneRow;
        hs[((size_t)t * BATCH + row) * (2 * HDIM) + (size_t)dir * HDIM + colh] = f2bf(hv);
      }
    }
    __syncthreads();
#pragma unroll
    for (int p4 = 0; p4 < 4; ++p4) {
      const int m = mArr[p4], nh = nhArr[p4];
#pragma unroll
      for (int r = 0; r < 8; ++r) {
        int row  = m * 16 + r + laneHi * 8;
        int colh = nh * 16 + laneRow;
        hb[(size_t)row * HDIM + colh] = f2bf(hstash[p4][r]);
      }
    }
    __syncthreads();
  }
}

// ---------------------------------------------------------------------------
// Kernel 5: emissions via WMMA: em = hs(32768x1024) @ plin(1024x16, 9 valid)
// 8 waves/block, one 16-row m-tile per wave, pipelined 32-iteration K loop.
// ---------------------------------------------------------------------------
__global__ __launch_bounds__(256)
void linear_wmma_kernel(const __bf16* __restrict__ hs,
                        const __bf16* __restrict__ plin,
                        const float* __restrict__ blin,
                        float* __restrict__ em) {
  const int lane = threadIdx.x & 31;
  const int laneRow = lane & 15, laneHi = lane >> 4;
  const int mtile = blockIdx.x * 8 + (threadIdx.x >> 5);

  v8f acc;
  float bv = (laneRow < NTAG) ? blin[laneRow] : 0.0f;
#pragma unroll
  for (int r = 0; r < 8; ++r) acc[r] = bv;

  const __bf16* arow = hs + (size_t)(mtile * 16 + laneRow) * (2 * HDIM);
  v16bf a = loadA(arow + laneHi * 8, arow + 16 + laneHi * 8);
  v16bf b = *(const v16bf*)(plin + (size_t)lane * 16);
  for (int kt = 0; kt < 32; ++kt) {
    v16bf an = a, bn = b;
    if (kt < 31) {
      int kb = (kt + 1) * 32;
      an = loadA(arow + kb + laneHi * 8, arow + kb + 16 + laneHi * 8);
      bn = *(const v16bf*)(plin + (size_t)(kt + 1) * 512 + (size_t)lane * 16);
    }
    acc = wmma_bf16(a, b, acc);
    a = an; b = bn;
  }

  if (laneRow < NTAG) {
#pragma unroll
    for (int r = 0; r < 8; ++r) {
      int row = mtile * 16 + r + laneHi * 8;   // row = s*64 + b
      int bb = row & 63, ss = row >> 6;
      em[((size_t)bb * S_LEN + ss) * NTAG + laneRow] = acc[r];
    }
  }
}

// ---------------------------------------------------------------------------
// Kernel 6: CRF numerator + forward algorithm. grid = B, block = 32 (1 wave).
// ---------------------------------------------------------------------------
__global__ void crf_kernel(const int* __restrict__ tags,
                           const float* __restrict__ em,
                           const float* __restrict__ start_tr,
                           const float* __restrict__ end_tr,
                           const float* __restrict__ trans,
                           float* __restrict__ scores) {
  const int b    = blockIdx.x;
  const int lane = threadIdx.x;
  const float NEG = -1e30f;

  float alpha = NEG;
  if (lane < NTAG) alpha = start_tr[lane] + em[((size_t)b * S_LEN) * NTAG + lane];
  const int tl = (lane < NTAG) ? lane : 0;

  for (int s = 1; s < S_LEN; ++s) {
    float e = (lane < NTAG) ? em[((size_t)b * S_LEN + s) * NTAG + lane] : 0.0f;
    float mx = NEG;
    float v[NTAG];
#pragma unroll
    for (int i = 0; i < NTAG; ++i) {
      float ai = __shfl(alpha, i, 32);
      v[i] = ai + trans[i * NTAG + tl];
      mx = fmaxf(mx, v[i]);
    }
    float sum = 0.0f;
#pragma unroll
    for (int i = 0; i < NTAG; ++i) sum += __expf(v[i] - mx);
    float na = mx + __logf(sum) + e;
    alpha = (lane < NTAG) ? na : NEG;
  }

  float fin = (lane < NTAG) ? (alpha + end_tr[lane]) : NEG;
  float mx = NEG;
#pragma unroll
  for (int i = 0; i < NTAG; ++i) mx = fmaxf(mx, __shfl(fin, i, 32));
  float sum = 0.0f;
#pragma unroll
  for (int i = 0; i < NTAG; ++i) sum += __expf(__shfl(fin, i, 32) - mx);
  float denom = mx + __logf(sum);

  if (lane == 0) {
    int prev = tags[b * S_LEN];
    float num = start_tr[prev] + em[((size_t)b * S_LEN) * NTAG + prev];
    for (int s = 1; s < S_LEN; ++s) {
      int tg = tags[b * S_LEN + s];
      num += em[((size_t)b * S_LEN + s) * NTAG + tg] + trans[prev * NTAG + tg];
      prev = tg;
    }
    num += end_tr[prev];
    scores[b] = num - denom;
  }
}

__global__ void reduce_kernel(const float* __restrict__ scores, float* __restrict__ out) {
  if (threadIdx.x == 0 && blockIdx.x == 0) {
    float s = 0.0f;
    for (int b = 0; b < BATCH; ++b) s += scores[b];
    out[0] = -s / (float)BATCH;
  }
}

// ---------------------------------------------------------------------------
// Launch
// ---------------------------------------------------------------------------
extern "C" void kernel_launch(void* const* d_in, const int* in_sizes, int n_in,
                              void* d_out, int out_size, void* d_ws, size_t ws_size,
                              hipStream_t stream) {
  (void)in_sizes; (void)n_in; (void)out_size;
  const int*   sentence = (const int*)  d_in[0];
  const int*   tags     = (const int*)  d_in[1];
  const float* emb      = (const float*)d_in[2];
  const float* wih_f    = (const float*)d_in[3];
  const float* whh_f    = (const float*)d_in[4];
  const float* bih_f    = (const float*)d_in[5];
  const float* bhh_f    = (const float*)d_in[6];
  const float* wih_b    = (const float*)d_in[7];
  const float* whh_b    = (const float*)d_in[8];
  const float* bih_b    = (const float*)d_in[9];
  const float* bhh_b    = (const float*)d_in[10];
  const float* wlin     = (const float*)d_in[11];
  const float* blin     = (const float*)d_in[12];
  const float* start_tr = (const float*)d_in[13];
  const float* end_tr   = (const float*)d_in[14];
  const float* trans    = (const float*)d_in[15];
  const float* h0       = (const float*)d_in[16];
  const float* c0       = (const float*)d_in[17];

  char* ws = (char*)d_ws;
  const size_t OFF_PW   = 0;
  const size_t OFF_X    = OFF_PW  + 2 * PW_PER_DIR * sizeof(__bf16);                    // + 8 MiB
  const size_t OFF_HS   = OFF_X   + (size_t)S_LEN * BATCH * EDIM * sizeof(__bf16);      // +32 MiB
  const size_t OFF_EM   = OFF_HS  + (size_t)S_LEN * BATCH * 2 * HDIM * sizeof(__bf16);  // +64 MiB
  const size_t OFF_SC   = OFF_EM  + (size_t)BATCH * S_LEN * NTAG * sizeof(float);
  const size_t OFF_PLIN = OFF_SC  + 256;
  const size_t OFF_PG   = (OFF_PLIN + 32 * 512 * sizeof(__bf16) + 255) & ~(size_t)255;
  const size_t PG_BYTES = 2 * (size_t)S_LEN * BATCH * GDIM * sizeof(__bf16);            // 256 MiB

  __bf16* pw   = (__bf16*)(ws + OFF_PW);
  __bf16* x    = (__bf16*)(ws + OFF_X);
  __bf16* hs   = (__bf16*)(ws + OFF_HS);
  float*  em   = (float*) (ws + OFF_EM);
  float*  sc   = (float*) (ws + OFF_SC);
  __bf16* plin = (__bf16*)(ws + OFF_PLIN);
  __bf16* pg   = (__bf16*)(ws + OFF_PG);

  const bool use_pg = (ws_size >= OFF_PG + PG_BYTES);

  {
    size_t n = 2 * PW_PER_DIR;
    pack_weights_kernel<<<(unsigned)((n + 255) / 256), 256, 0, stream>>>(
        wih_f, whh_f, wih_b, whh_b, pw);
  }
  pack_wlin_kernel<<<64, 256, 0, stream>>>(wlin, plin);
  {
    size_t n = (size_t)S_LEN * BATCH * EDIM;
    embed_cast_kernel<<<(unsigned)((n + 255) / 256), 256, 0, stream>>>(sentence, emb, x);
  }

  if (use_pg) {
    pregate_kernel<<<8192, 256, 0, stream>>>(x, pw, bih_f, bhh_f, bih_b, bhh_b, pg);
    lstm_pg_kernel<<<2, 1024, 0, stream>>>(h0, c0, pw, pg, hs);
  } else {
    lstm_fused_kernel<<<2, 1024, 0, stream>>>(h0, c0, bih_f, bhh_f, bih_b, bhh_b, pw, x, hs);
  }

  linear_wmma_kernel<<<256, 256, 0, stream>>>(hs, plin, blin, em);
  crf_kernel<<<BATCH, 32, 0, stream>>>(tags, em, start_tr, end_tr, trans, sc);
  reduce_kernel<<<1, 64, 0, stream>>>(sc, (float*)d_out);
}
